// seqsim_atn_76716705841641
// MI455X (gfx1250) — matrix operand order repaired and checked by reading, not compiled, not run on hardware
//
#include <hip/hip_runtime.h>
#include <hip/hip_bf16.h>
#include <stdint.h>

// Problem constants (match reference)
#define BQ     8
#define DBYTES 128
#define RREF   4096
#define SEDGE  32768
#define NSEG   1024
#define OUTD   512
#define DBITS  1024

typedef __attribute__((ext_vector_type(16))) __bf16          v16bf;
typedef __attribute__((ext_vector_type(16))) unsigned short  v16u;
typedef __attribute__((ext_vector_type(8)))  unsigned short  v8u;
typedef __attribute__((ext_vector_type(8)))  float           v8f;

// ---------------------------------------------------------------------------
// Kernel 1: dists[b*R + r] = popcnt(q & refs[r]) / popcnt(q_ok & ok_pos[r])
// ---------------------------------------------------------------------------
__global__ __launch_bounds__(256) void k_dists(const uint8_t* __restrict__ Q,
                                               const uint8_t* __restrict__ Qok,
                                               const uint8_t* __restrict__ refs,
                                               const uint8_t* __restrict__ okpos,
                                               float* __restrict__ dists) {
    int idx = blockIdx.x * blockDim.x + threadIdx.x;   // b*RREF + r
    if (idx >= BQ * RREF) return;
    int b = idx >> 12;
    int r = idx & (RREF - 1);
    const uint32_t* q  = (const uint32_t*)(Q     + b * DBYTES);
    const uint32_t* qo = (const uint32_t*)(Qok   + b * DBYTES);
    const uint32_t* rf = (const uint32_t*)(refs  + r * DBYTES);
    const uint32_t* op = (const uint32_t*)(okpos + r * DBYTES);
    int match = 0, ok = 0;
#pragma unroll
    for (int i = 0; i < DBYTES / 4; ++i) {
        match += __popc(q[i]  & rf[i]);
        ok    += __popc(qo[i] & op[i]);
    }
    dists[idx] = (float)match / (float)ok;
}

// ---------------------------------------------------------------------------
// Kernel 2: CSR offsets for sorted segment ids: segoff[n] = lower_bound(n)
// ---------------------------------------------------------------------------
__global__ __launch_bounds__(256) void k_segoff(const int* __restrict__ segments,
                                                int* __restrict__ segoff) {
    int n = blockIdx.x * blockDim.x + threadIdx.x;
    if (n > NSEG) return;
    int lo = 0, hi = SEDGE;
    while (lo < hi) {
        int mid = (lo + hi) >> 1;
        if (segments[mid] < n) lo = mid + 1; else hi = mid;
    }
    segoff[n] = lo;
}

// ---------------------------------------------------------------------------
// Kernel 3: W_o (1024x512 f32, row-major) -> Wt (512x1024 bf16, TRANSPOSED)
// Coalesced reads; scattered 2B writes (1 MB total, trivial).
// ---------------------------------------------------------------------------
__global__ __launch_bounds__(256) void k_wbf(const float* __restrict__ W,
                                             unsigned short* __restrict__ Wt) {
    int i = blockIdx.x * blockDim.x + threadIdx.x;
    if (i >= DBITS * OUTD) return;
    int k = i >> 9;          // row in W   (0..1023)
    int o = i & (OUTD - 1);  // col in W   (0..511)
    __bf16 h = (__bf16)W[i]; // RNE f32 -> bf16
    Wt[o * DBITS + k] = __builtin_bit_cast(unsigned short, h);
}

// ---------------------------------------------------------------------------
// Kernel 4a: unpack refs bits -> Abf (4096x1024 bf16 row-major, 0.0/1.0)
// numpy unpackbits is MSB-first within each byte. One thread per dword.
// ---------------------------------------------------------------------------
__global__ __launch_bounds__(256) void k_unpackA(const uint8_t* __restrict__ refs,
                                                 unsigned short* __restrict__ Abf) {
    int idx = blockIdx.x * blockDim.x + threadIdx.x;   // r*32 + dword
    if (idx >= RREF * (DBYTES / 4)) return;
    int r = idx >> 5;
    int d = idx & 31;
    uint32_t w = ((const uint32_t*)(refs + r * DBYTES))[d];
    v16u o0, o1;
#pragma unroll
    for (int j = 0; j < 16; ++j) {
        int bit0 = (w >> (((j >> 3) << 3) + 7 - (j & 7))) & 1;
        int j2 = j + 16;
        int bit1 = (w >> (((j2 >> 3) << 3) + 7 - (j2 & 7))) & 1;
        o0[j] = bit0 ? (unsigned short)0x3F80 : (unsigned short)0;
        o1[j] = bit1 ? (unsigned short)0x3F80 : (unsigned short)0;
    }
    unsigned short* dst = Abf + r * DBITS + d * 32;
    *(v16u*)(dst)      = o0;
    *(v16u*)(dst + 16) = o1;
}

// ---------------------------------------------------------------------------
// Kernel 4b: P = Abf @ Wt^T  via v_wmma_f32_16x16x32_bf16
//   One wave per 16x64 output tile: 4 accumulators share each A fragment.
//   Per K-step (K=32): 2x b128 A loads, 4x 32B B loads, 4 WMMAs.
//   Lane layouts used by this kernel (16-bit operands, wave32):
//     A: row m=lane&15; elems 0..7  = K[16*half .. 16*half+7]
//                       elems 8..15 = K[16*half+8 .. 16*half+15]
//     B: col n=lane&15; elem e -> K = e + 16*half (contiguous 16)
//     C/D f32: col = lane&15; VGPR g -> row = g + 8*half
// ---------------------------------------------------------------------------
__global__ __launch_bounds__(32) void k_pgemm(const unsigned short* __restrict__ Abf,
                                              const unsigned short* __restrict__ Wt,
                                              float* __restrict__ P) {
    const int lane = threadIdx.x;
    const int half = lane >> 4;
    const int m    = lane & 15;
    const int tr = blockIdx.x * 16;   // row tile in R
    const int tc = blockIdx.y * 64;   // col tile in OUTD

    const unsigned short* arow = Abf + (tr + m) * DBITS;
    const unsigned short* b0 = Wt + (tc +  0 + m) * DBITS + (half << 4);
    const unsigned short* b1 = Wt + (tc + 16 + m) * DBITS + (half << 4);
    const unsigned short* b2 = Wt + (tc + 32 + m) * DBITS + (half << 4);
    const unsigned short* b3 = Wt + (tc + 48 + m) * DBITS + (half << 4);

    v8f acc0 = {}, acc1 = {}, acc2 = {}, acc3 = {};
    for (int kb = 0; kb < DBITS / 32; ++kb) {
        const int k0 = kb * 32;
        // A fragment: two contiguous 16B chunks
        v8u lo = *(const v8u*)(arow + k0 + (half << 4));
        v8u hi = *(const v8u*)(arow + k0 + 8 + (half << 4));
        v16u au = __builtin_shufflevector(lo, hi, 0, 1, 2, 3, 4, 5, 6, 7,
                                                   8, 9, 10, 11, 12, 13, 14, 15);
        v16bf av = __builtin_bit_cast(v16bf, au);
        // B fragments: one contiguous 32B chunk per column tile
        v16bf bv0 = __builtin_bit_cast(v16bf, *(const v16u*)(b0 + k0));
        v16bf bv1 = __builtin_bit_cast(v16bf, *(const v16u*)(b1 + k0));
        v16bf bv2 = __builtin_bit_cast(v16bf, *(const v16u*)(b2 + k0));
        v16bf bv3 = __builtin_bit_cast(v16bf, *(const v16u*)(b3 + k0));
        acc0 = __builtin_amdgcn_wmma_f32_16x16x32_bf16(false, av, false, bv0, (short)0, acc0, false, false);
        acc1 = __builtin_amdgcn_wmma_f32_16x16x32_bf16(false, av, false, bv1, (short)0, acc1, false, false);
        acc2 = __builtin_amdgcn_wmma_f32_16x16x32_bf16(false, av, false, bv2, (short)0, acc2, false, false);
        acc3 = __builtin_amdgcn_wmma_f32_16x16x32_bf16(false, av, false, bv3, (short)0, acc3, false, false);
    }
    float* prow = P + (tr + (half << 3)) * OUTD + tc + m;
#pragma unroll
    for (int g = 0; g < 8; ++g) {
        prow[g * OUTD +  0] = acc0[g];
        prow[g * OUTD + 16] = acc1[g];
        prow[g * OUTD + 32] = acc2[g];
        prow[g * OUTD + 48] = acc3[g];
    }
}

// ---------------------------------------------------------------------------
// Kernel 5: per (b, n): segment softmax over edges, then
//   out[b,n,:] = b_o + sum_s score_s * P[ref2seg[s], :]
// ---------------------------------------------------------------------------
__global__ __launch_bounds__(256) void k_attn(const int* __restrict__ ref2seg,
                                              const int* __restrict__ segoff,
                                              const float* __restrict__ dists,
                                              const float* __restrict__ P,
                                              const float* __restrict__ b_o,
                                              float* __restrict__ out) {
    const int bid = blockIdx.x;            // b*NSEG + n
    const int b = bid >> 10;
    const int n = bid & (NSEG - 1);
    const int start = segoff[n];
    const int end   = segoff[n + 1];
    const int t = threadIdx.x;

    __shared__ float red[256];
    const float* db = dists + b * RREF;

    // pass 1: segment max
    float mx = -INFINITY;
    for (int s = start + t; s < end; s += 256) mx = fmaxf(mx, db[ref2seg[s]]);
    red[t] = mx; __syncthreads();
    for (int off = 128; off > 0; off >>= 1) {
        if (t < off) red[t] = fmaxf(red[t], red[t + off]);
        __syncthreads();
    }
    mx = red[0]; __syncthreads();

    // pass 2: segment sum of exp(d - mx)
    float z = 0.0f;
    for (int s = start + t; s < end; s += 256) z += expf(db[ref2seg[s]] - mx);
    red[t] = z; __syncthreads();
    for (int off = 128; off > 0; off >>= 1) {
        if (t < off) red[t] += red[t + off];
        __syncthreads();
    }
    z = red[0];
    const float invz = (end > start) ? (1.0f / z) : 0.0f;

    // pass 3: weighted sum of P rows (each thread owns outputs t and t+256)
    float acc0 = b_o[t];
    float acc1 = b_o[t + 256];
    for (int s = start; s < end; ++s) {
        int r = ref2seg[s];
        float sc = expf(db[r] - mx) * invz;
        const float* pr = P + r * OUTD;
        acc0 = fmaf(sc, pr[t],       acc0);
        acc1 = fmaf(sc, pr[t + 256], acc1);
    }
    float* o = out + (long)bid * OUTD;
    o[t]       = acc0;
    o[t + 256] = acc1;
}

// ---------------------------------------------------------------------------
// Launch
// ---------------------------------------------------------------------------
extern "C" void kernel_launch(void* const* d_in, const int* in_sizes, int n_in,
                              void* d_out, int out_size, void* d_ws, size_t ws_size,
                              hipStream_t stream) {
    const uint8_t* Q      = (const uint8_t*)d_in[0];
    const uint8_t* Qok    = (const uint8_t*)d_in[1];
    const uint8_t* refs   = (const uint8_t*)d_in[2];
    const uint8_t* okpos  = (const uint8_t*)d_in[3];
    const int*     r2s    = (const int*)d_in[4];
    const int*     segs   = (const int*)d_in[5];
    // d_in[6] = N (compile-time constant NSEG)
    const float*   W_o    = (const float*)d_in[7];
    const float*   b_o    = (const float*)d_in[8];
    float*         out    = (float*)d_out;

    // Workspace layout (256-byte aligned chunks)
    uint8_t* ws = (uint8_t*)d_ws;
    float*          dists  = (float*)(ws);                        // 131072 B
    int*            segoff = (int*)(ws + 131072);                 // 4100 B (pad->4352)
    unsigned short* Wt     = (unsigned short*)(ws + 135424);      // 1048576 B
    unsigned short* Abf    = (unsigned short*)(ws + 1184000);     // 8388608 B
    float*          P      = (float*)(ws + 9572608);              // 8388608 B
    // total = 17961216 B (~17.1 MiB)

    k_dists  <<<(BQ * RREF + 255) / 256, 256, 0, stream>>>(Q, Qok, refs, okpos, dists);
    k_segoff <<<(NSEG + 1 + 255) / 256, 256, 0, stream>>>(segs, segoff);
    k_wbf    <<<(DBITS * OUTD) / 256, 256, 0, stream>>>(W_o, Wt);
    k_unpackA<<<(RREF * (DBYTES / 4)) / 256, 256, 0, stream>>>(refs, Abf);
    k_pgemm  <<<dim3(RREF / 16, OUTD / 64), 32, 0, stream>>>(Abf, Wt, P);
    k_attn   <<<BQ * NSEG, 256, 0, stream>>>(r2s, segoff, dists, P, b_o, out);
}
